// HamburgerV2Plus_29961691857142
// MI455X (gfx1250) — matrix operand order repaired and checked by reading, not compiled, not run on hardware
//
#include <hip/hip_runtime.h>
#include <math.h>
#include <stdint.h>

// ---------------------------------------------------------------------------
// HamburgerV2Plus forward for MI455X (gfx1250, wave32).
// GEMMs: bf16 operands in global memory, fragment-major LDS tiles (double
// buffered), async global->LDS copies where contiguity allows, and
// v_wmma_f32_16x16x32_bf16 compute (4 per wave per K-step).
// ---------------------------------------------------------------------------

typedef __attribute__((ext_vector_type(16))) __bf16          v16bf;
typedef __attribute__((ext_vector_type(16))) unsigned short  v16us;
typedef __attribute__((ext_vector_type(8)))  float           v8f;

union FragBF {
    v16bf v;
    v16us us;
};

__device__ __forceinline__ unsigned short f32_to_bf16(float f) {
    unsigned int u = __float_as_uint(f);
    unsigned int r = u + 0x7FFFu + ((u >> 16) & 1u); // round-to-nearest-even
    return (unsigned short)(r >> 16);
}

// Async 16-byte global -> LDS copy (CDNA5, tracked by ASYNCcnt).
__device__ __forceinline__ void async_b128(unsigned short* lds_dst,
                                           const unsigned short* gsrc) {
    unsigned loff = (unsigned)(uintptr_t)lds_dst; // low 32 bits = LDS offset
    asm volatile("global_load_async_to_lds_b128 %0, %1, off"
                 :: "v"(loff), "v"(gsrc)
                 : "memory");
}

__device__ __forceinline__ void wait_async0() {
    asm volatile("s_wait_asynccnt 0x0" ::: "memory");
}

#define EPS 1e-6f

// ---------------------------------------------------------------------------
// Generic strided batched GEMM: C[b][m,n] = sum_k A[b][m,k]*B[b][k,n]
// A,B are bf16 (ushort). Output to Cf (f32) and/or Ch (bf16).
// Exactly one of {sAr,sAc} and one of {sBr,sBc} must be 1.
// Block tile 128(M) x 64(N), K step 32. 256 threads = 8 waves,
// wave (mwv = wave>>1, nwv = wave&1) computes 32x32 via 4 WMMAs.
// LDS fragment-major layout: frag[sub][lane][16 halves].
// ---------------------------------------------------------------------------
__global__ __launch_bounds__(256)
void k_wmma_gemm(const unsigned short* __restrict__ A,
                 const unsigned short* __restrict__ Bm,
                 float* __restrict__ Cf, unsigned short* __restrict__ Ch,
                 const float* __restrict__ bias,
                 int M, int N, int K,
                 long long bA, long long sAr, long long sAc,
                 long long bB, long long sBr, long long sBc,
                 long long bC, int relu)
{
    __shared__ __align__(32) unsigned short Af[2 * 8 * 32 * 16]; // 16 KB
    __shared__ __align__(32) unsigned short Bf[2 * 4 * 32 * 16]; //  8 KB

    const int tid  = threadIdx.x;
    const int wave = tid >> 5;
    const int lane = tid & 31;
    const int half = lane >> 4;
    const int l16  = lane & 15;
    const int mwv  = wave >> 1;   // 4 M-groups of 32 rows
    const int nwv  = wave & 1;    // 2 N-groups of 32 cols

    const int m0 = blockIdx.y * 128;
    const int n0 = blockIdx.x * 64;
    const long long batch = blockIdx.z;
    const unsigned short* Ab = A  + batch * bA;
    const unsigned short* Bb = Bm + batch * bB;

    const bool fastA = (sAc == 1);  // K contiguous in A -> async copy
    const bool fastB = (sBr == 1);  // K contiguous in B -> async copy

    // ---- per-thread staging descriptors (A: 2 chunks of 8, B: 1 chunk) ----
    const unsigned short* pa[2];
    int da[2];
    #pragma unroll
    for (int j = 0; j < 2; ++j) {
        int qa = tid + 256 * j;
        if (fastA) {
            // chunk = row r, 8 consecutive K at c0
            int r  = qa >> 2;
            int c0 = (qa & 3) * 8;
            int rg = m0 + r; if (rg > M - 1) rg = M - 1;
            int hf = (c0 & 8) ? 1 : 0;
            int i0 = (c0 & 16) ? 8 : 0;
            da[j] = (r >> 4) * 512 + (hf * 16 + (r & 15)) * 16 + i0;
            pa[j] = Ab + (long long)rg * sAr + c0;
        } else {
            // chunk = K col c, 8 consecutive rows at r0 (sAr == 1)
            int c  = qa >> 4;
            int r0 = (qa & 15) * 8;
            int rg = m0 + r0; if (rg > M - 8) rg = M - 8;
            int hf = (c & 8) ? 1 : 0;
            int i0 = (c & 7) + ((c & 16) ? 8 : 0);
            da[j] = (r0 >> 4) * 512 + (hf * 16 + (r0 & 15)) * 16 + i0;
            pa[j] = Ab + rg + (long long)c * sAc;
        }
    }
    const long long stepA = fastA ? 32 : 32 * sAc;

    const unsigned short* pb;
    int db;
    {
        int qb = tid;
        if (fastB) {
            // chunk = col n, 8 consecutive K at kc0 (sBr == 1)
            int n   = qb >> 2;
            int kc0 = (qb & 3) * 8;
            int hf  = (kc0 & 8) ? 1 : 0;
            int i0  = (kc0 & 16) ? 8 : 0;
            db = (n >> 4) * 512 + (hf * 16 + (n & 15)) * 16 + i0;
            pb = Bb + kc0 + (long long)(n0 + n) * sBc;
        } else {
            // chunk = K row kr, 8 consecutive cols at nc0 (sBc == 1)
            int kr  = qb >> 3;
            int nc0 = (qb & 7) * 8;
            int hf  = (kr & 8) ? 1 : 0;
            int i0  = (kr & 7) + ((kr & 16) ? 8 : 0);
            db = (nc0 >> 4) * 512 + (hf * 16 + (nc0 & 15)) * 16 + i0;
            pb = Bb + (long long)kr * sBr + (n0 + nc0);
        }
    }
    const long long stepB = fastB ? 32 : 32 * sBr;

    // ---- staging: async b128 per operand when K-contiguous, else scatter ----
    auto stage = [&](int sel) {
        unsigned short* AfS = Af + sel * 4096;
        unsigned short* BfS = Bf + sel * 2048;
        if (fastA) {
            async_b128(AfS + da[0], pa[0]);
            async_b128(AfS + da[1], pa[1]);
        } else {
            #pragma unroll
            for (int j = 0; j < 2; ++j) {
                union { uint4 q; unsigned short s[8]; } t;
                t.q = *(const uint4*)pa[j];
                unsigned short* d = AfS + da[j];
                #pragma unroll
                for (int e = 0; e < 8; ++e) d[e * 16] = t.s[e];
            }
        }
        pa[0] += stepA; pa[1] += stepA;
        if (fastB) {
            async_b128(BfS + db, pb);
        } else {
            union { uint4 q; unsigned short s[8]; } t;
            t.q = *(const uint4*)pb;
            unsigned short* d = BfS + db;
            #pragma unroll
            for (int e = 0; e < 8; ++e) d[e * 16] = t.s[e];
        }
        pb += stepB;
    };

    v8f acc[2][2] = {{{}, {}}, {{}, {}}};

    stage(0);
    wait_async0();
    __syncthreads();

    const int nT = K >> 5;
    for (int t = 0; t < nT; ++t) {
        const int cur = t & 1;
        if (t + 1 < nT) stage(cur ^ 1); // overlap copy of next tile

        const v16us* afp = (const v16us*)(Af + cur * 4096 + (mwv * 2) * 512 + lane * 16);
        const v16us* bfp = (const v16us*)(Bf + cur * 2048 + (nwv * 2) * 512 + lane * 16);
        FragBF a0, a1, b0, b1;
        a0.us = afp[0];
        a1.us = afp[32];   // next m-subtile (+512 ushorts)
        b0.us = bfp[0];
        b1.us = bfp[32];   // next n-subtile

        acc[0][0] = __builtin_amdgcn_wmma_f32_16x16x32_bf16(
                        false, a0.v, false, b0.v, (short)0, acc[0][0], false, false);
        acc[0][1] = __builtin_amdgcn_wmma_f32_16x16x32_bf16(
                        false, a0.v, false, b1.v, (short)0, acc[0][1], false, false);
        acc[1][0] = __builtin_amdgcn_wmma_f32_16x16x32_bf16(
                        false, a1.v, false, b0.v, (short)0, acc[1][0], false, false);
        acc[1][1] = __builtin_amdgcn_wmma_f32_16x16x32_bf16(
                        false, a1.v, false, b1.v, (short)0, acc[1][1], false, false);

        wait_async0();
        __syncthreads();
    }

    // ---- epilogue: C/D layout VGPR j -> M = j + 8*half, N = lane&15 ----
    #pragma unroll
    for (int am = 0; am < 2; ++am) {
        #pragma unroll
        for (int bn = 0; bn < 2; ++bn) {
            #pragma unroll
            for (int j = 0; j < 8; ++j) {
                int gr = m0 + (mwv * 2 + am) * 16 + half * 8 + j;
                int gc = n0 + (nwv * 2 + bn) * 16 + l16;
                if (gr < M) {
                    float v = acc[am][bn][j];
                    if (bias) v += bias[gr];
                    if (relu) v = fmaxf(v, 0.0f);
                    long long idx = batch * bC + (long long)gr * N + gc;
                    if (Cf) Cf[idx] = v;
                    if (Ch) Ch[idx] = f32_to_bf16(v);
                }
            }
        }
    }
}

// ---------------------------------------------------------------------------
// f32 -> bf16 bulk convert
// ---------------------------------------------------------------------------
__global__ __launch_bounds__(256)
void k_convert_bf16(const float* __restrict__ src, unsigned short* __restrict__ dst,
                    long long n)
{
    long long i = (long long)blockIdx.x * blockDim.x + threadIdx.x;
    if (i < n) dst[i] = f32_to_bf16(src[i]);
}

// ---------------------------------------------------------------------------
// bases = l2norm(bases0) over D axis; writes f32 + bf16 mirror.
// ---------------------------------------------------------------------------
__global__ __launch_bounds__(256)
void k_l2norm(const float* __restrict__ b0, float* __restrict__ bout,
              unsigned short* __restrict__ bout16, int Bn, int D, int R)
{
    int col  = blockIdx.x * (blockDim.x >> 5) + (threadIdx.x >> 5);
    int lane = threadIdx.x & 31;
    if (col >= Bn * R) return;
    int b = col / R, r = col % R;
    const float* p = b0 + (long long)b * D * R + r;
    float s = 0.0f;
    for (int d = lane; d < D; d += 32) { float v = p[(long long)d * R]; s += v * v; }
    #pragma unroll
    for (int m = 16; m; m >>= 1) s += __shfl_xor(s, m, 32);
    float inv = 1.0f / fmaxf(sqrtf(s), 1e-12f);
    long long base = (long long)b * D * R + r;
    for (int d = lane; d < D; d += 32) {
        float v = p[(long long)d * R] * inv;
        bout[base + (long long)d * R]   = v;
        bout16[base + (long long)d * R] = f32_to_bf16(v);
    }
}

// ---------------------------------------------------------------------------
// Softmax over R=64 (last axis); writes f32 + bf16 mirror.
// ---------------------------------------------------------------------------
__global__ __launch_bounds__(256)
void k_softmax64(const float* __restrict__ in, float* __restrict__ out,
                 unsigned short* __restrict__ out16, long long rows)
{
    long long row = (long long)blockIdx.x * (blockDim.x >> 5) + (threadIdx.x >> 5);
    int lane = threadIdx.x & 31;
    if (row >= rows) return;
    const float* p = in + row * 64;
    float a = p[lane], b = p[lane + 32];
    float mx = fmaxf(a, b);
    #pragma unroll
    for (int m = 16; m; m >>= 1) mx = fmaxf(mx, __shfl_xor(mx, m, 32));
    float ea = __expf(a - mx), eb = __expf(b - mx);
    float s = ea + eb;
    #pragma unroll
    for (int m = 16; m; m >>= 1) s += __shfl_xor(s, m, 32);
    float inv = 1.0f / s;
    float va = ea * inv, vb = eb * inv;
    out[row * 64 + lane]        = va;
    out[row * 64 + lane + 32]   = vb;
    out16[row * 64 + lane]      = f32_to_bf16(va);
    out16[row * 64 + lane + 32] = f32_to_bf16(vb);
}

// ---------------------------------------------------------------------------
// Multiplicative NMF update: c *= num/(den+EPS); writes f32 + bf16 mirror.
// ---------------------------------------------------------------------------
__global__ __launch_bounds__(256)
void k_mulupdate(float* __restrict__ c, unsigned short* __restrict__ c16,
                 const float* __restrict__ num, const float* __restrict__ den,
                 long long n)
{
    long long i = (long long)blockIdx.x * blockDim.x + threadIdx.x;
    if (i < n) {
        float v = c[i] * num[i] / (den[i] + EPS);
        c[i]   = v;
        c16[i] = f32_to_bf16(v);
    }
}

// ---------------------------------------------------------------------------
// Residual epilogue: out = relu(coef_ham*out + coef_shortcut*x)
// ---------------------------------------------------------------------------
__global__ __launch_bounds__(256)
void k_final(float* __restrict__ out, const float* __restrict__ x,
             const float* __restrict__ cs, const float* __restrict__ ch,
             long long n)
{
    long long i = (long long)blockIdx.x * blockDim.x + threadIdx.x;
    if (i < n) out[i] = fmaxf(ch[0] * out[i] + cs[0] * x[i], 0.0f);
}

// ---------------------------------------------------------------------------
// Host-side orchestration
// ---------------------------------------------------------------------------
static inline void launch_gemm(hipStream_t s,
                               const unsigned short* A, const unsigned short* B,
                               float* Cf, unsigned short* Ch,
                               int M, int N, int K,
                               long long bA, long long sAr, long long sAc,
                               long long bB, long long sBr, long long sBc,
                               long long bC, const float* bias, int relu, int batch)
{
    dim3 grid((N + 63) / 64, (M + 127) / 128, batch);
    k_wmma_gemm<<<grid, 256, 0, s>>>(A, B, Cf, Ch, bias, M, N, K,
                                     bA, sAr, sAc, bB, sBr, sBc, bC, relu);
}

extern "C" void kernel_launch(void* const* d_in, const int* in_sizes, int n_in,
                              void* d_out, int out_size, void* d_ws, size_t ws_size,
                              hipStream_t stream)
{
    (void)n_in; (void)ws_size; (void)out_size;

    const float* x        = (const float*)d_in[0]; // [B,512,64,64]
    const float* bases0   = (const float*)d_in[1]; // [B,512,64]
    const float* w_lower  = (const float*)d_in[2]; // [512,512]
    const float* b_lower  = (const float*)d_in[3];
    const float* w_cheese = (const float*)d_in[4];
    const float* b_cheese = (const float*)d_in[5];
    const float* w_upper  = (const float*)d_in[6];
    const float* c_short  = (const float*)d_in[7];
    const float* c_ham    = (const float*)d_in[8];
    float* out            = (float*)d_out;

    const int C = 512, R = 64;
    const int B = in_sizes[1] / (C * R);   // 8
    const int N = in_sizes[0] / (B * C);   // 4096

    const long long szX    = (long long)C * N;
    const long long szCoef = (long long)N * R;
    const long long szBas  = (long long)C * R;
    const long long szRR   = (long long)R * R;
    const long long szW    = (long long)C * C;

    typedef unsigned short u16;
    char* p = (char*)d_ws;
    u16*   xb    = (u16*)p;   p += (size_t)B * szX    * 2;  // 32 MB (reused for recon out)
    u16*   hb    = (u16*)p;   p += (size_t)B * szX    * 2;  // 32 MB (reused for cheese out)
    float* coef  = (float*)p; p += (size_t)B * szCoef * 4;  //  8 MB
    u16*   coefb = (u16*)p;   p += (size_t)B * szCoef * 2;  //  4 MB
    float* xtb   = (float*)p; p += (size_t)B * szCoef * 4;  //  8 MB
    float* den   = (float*)p; p += (size_t)B * szCoef * 4;  //  8 MB
    float* xc    = (float*)p; p += (size_t)B * szBas  * 4;  //  1 MB
    float* bas   = (float*)p; p += (size_t)B * szBas  * 4;  //  1 MB
    u16*   basb  = (u16*)p;   p += (size_t)B * szBas  * 2;  // 0.5 MB
    u16*   btbb  = (u16*)p;   p += (size_t)B * szRR   * 2;  // 64 KB
    u16*   ctcb  = (u16*)p;   p += (size_t)B * szRR   * 2;  // 64 KB
    u16*   wlb   = (u16*)p;   p += (size_t)szW * 2;         // 512 KB
    u16*   wcb   = (u16*)p;   p += (size_t)szW * 2;
    u16*   wub   = (u16*)p;   p += (size_t)szW * 2;

    // 0) one-time bf16 conversions
    {
        long long n = (long long)B * szX;
        k_convert_bf16<<<(unsigned)((n + 255) / 256), 256, 0, stream>>>(x, xb, n);
        k_convert_bf16<<<(unsigned)((szW + 255) / 256), 256, 0, stream>>>(w_lower,  wlb, szW);
        k_convert_bf16<<<(unsigned)((szW + 255) / 256), 256, 0, stream>>>(w_cheese, wcb, szW);
        k_convert_bf16<<<(unsigned)((szW + 255) / 256), 256, 0, stream>>>(w_upper,  wub, szW);
    }

    // 1) bases = l2norm(bases0)
    k_l2norm<<<(B * R + 7) / 8, 256, 0, stream>>>(bases0, bas, basb, B, C, R);

    // 2) lower bread: hb = bf16(relu(w_lower @ x + b_lower))
    launch_gemm(stream, wlb, xb, nullptr, hb, C, N, C,
                0, C, 1,  szX, N, 1,  szX, b_lower, 1, B);

    auto gemm_xtb = [&]() { // xtb[n,r] = sum_d h[d,n]*bas[d,r]
        launch_gemm(stream, hb, basb, xtb, nullptr, N, R, C,
                    szX, 1, N,  szBas, R, 1,  szCoef, nullptr, 0, B);
    };
    auto gemm_btb = [&]() { // btb[r,s] = sum_d bas[d,r]*bas[d,s]  (bf16 out)
        launch_gemm(stream, basb, basb, nullptr, btbb, R, R, C,
                    szBas, 1, R,  szBas, R, 1,  szRR, nullptr, 0, B);
    };
    auto gemm_coef_den = [&]() { // den[n,s] = sum_r coef[n,r]*btb[r,s]
        launch_gemm(stream, coefb, btbb, den, nullptr, N, R, R,
                    szCoef, R, 1,  szRR, R, 1,  szCoef, nullptr, 0, B);
    };
    auto coef_update = [&]() {
        long long n = (long long)B * szCoef;
        k_mulupdate<<<(unsigned)((n + 255) / 256), 256, 0, stream>>>(coef, coefb, xtb, den, n);
    };

    // 3) initial coef = softmax(x^T bases)
    gemm_xtb();
    k_softmax64<<<(unsigned)(((long long)B * N + 7) / 8), 256, 0, stream>>>(
        xtb, coef, coefb, (long long)B * N);

    // 4) 6 multiplicative-update steps
    for (int step = 0; step < 6; ++step) {
        gemm_xtb();
        gemm_btb();
        gemm_coef_den();
        coef_update();
        // xc[d,r] = sum_n h[d,n]*coef[n,r]
        launch_gemm(stream, hb, coefb, xc, nullptr, C, R, N,
                    szX, N, 1,  szCoef, R, 1,  szBas, nullptr, 0, B);
        // ctc[r,s] = sum_n coef[n,r]*coef[n,s]  (bf16 out)
        launch_gemm(stream, coefb, coefb, nullptr, ctcb, R, R, N,
                    szCoef, 1, R,  szCoef, R, 1,  szRR, nullptr, 0, B);
        // den[d,s] = sum_r bas[d,r]*ctc[r,s]
        launch_gemm(stream, basb, ctcb, den, nullptr, C, R, R,
                    szBas, R, 1,  szRR, R, 1,  szBas, nullptr, 0, B);
        long long nb = (long long)B * szBas;
        k_mulupdate<<<(unsigned)((nb + 255) / 256), 256, 0, stream>>>(bas, basb, xc, den, nb);
    }

    // 5) differentiable final coef refinement
    gemm_xtb();
    gemm_btb();
    gemm_coef_den();
    coef_update();

    // 6) reconstruction: xb[d,n] = bf16(sum_r bas[d,r]*coef[n,r])  (B fast: K contig)
    launch_gemm(stream, basb, coefb, nullptr, xb, C, N, R,
                szBas, R, 1,  szCoef, 1, R,  szX, nullptr, 0, B);

    // 7) cheese: hb = bf16(relu(w_cheese @ recon + b_cheese))
    launch_gemm(stream, wcb, xb, nullptr, hb, C, N, C,
                0, C, 1,  szX, N, 1,  szX, b_cheese, 1, B);

    // 8) upper bread: out = w_upper @ cheese (f32, no bias)
    launch_gemm(stream, wub, hb, out, nullptr, C, N, C,
                0, C, 1,  szX, N, 1,  szX, nullptr, 0, B);

    // 9) residual: out = relu(coef_ham*out + coef_shortcut*x)
    {
        long long n = (long long)B * szX;
        k_final<<<(unsigned)((n + 255) / 256), 256, 0, stream>>>(out, x, c_short, c_ham, n);
    }
}
